// EfficientMixedScoreMultiHeadAttention_6700148982477
// MI455X (gfx1250) — matrix operand order repaired and checked by reading, not compiled, not run on hardware
//
#include <hip/hip_runtime.h>
#include <math.h>

// EfficientMixedScoreMultiHeadAttention for MI455X (gfx1250), wave32 + WMMA bf16.
// B=4, R=C=512, E=256, H=16, D=16. All GEMM-shaped work on v_wmma_f32_16x16x32_bf16.
// All weights pre-converted to bf16 once; activations converted at load.

#define BB 4
#define RR 512
#define CC 512
#define EE 256
#define HH 16
#define DD 16

typedef __attribute__((ext_vector_type(16))) __bf16 v16bf;
typedef __attribute__((ext_vector_type(8)))  float  v8f;

__device__ __forceinline__ v8f wmma_bf16(v16bf a, v16bf b, v8f c) {
  // (neg_a, A, neg_b, B, c_mod, C, reuse_a, reuse_b)
  return __builtin_amdgcn_wmma_f32_16x16x32_bf16(false, a, false, b, (short)0, c, false, false);
}

// CDNA5 wave32 WMMA VGPR layouts (CDNA5 ISA):
// A (16x32 bf16): lane holds row m=lane&15; element e -> K = (lane<16?0:8) + (e<8?e:e+8)
// B (32x16 bf16): lane holds col n=lane&15; element e -> K = (lane<16?0:8) + (e<8?e:e+8)
// C (16x16 f32):  element i -> row = (lane<16?0:8)+i, col = lane&15
__device__ __forceinline__ int a_k(int lane, int e) { return (lane < 16 ? 0 : 8) + (e < 8 ? e : e + 8); }
__device__ __forceinline__ int b_k(int lane, int e) { return (lane < 16 ? 0 : 8) + (e < 8 ? e : e + 8); }
__device__ __forceinline__ int c_row(int lane, int i) { return (lane < 16 ? 0 : 8) + i; }

// ---------------------------------------------------------------------------
// Kernel 0a: f32 -> bf16 weight conversion (run once per weight matrix)
__global__ __launch_bounds__(256)
void f32_to_bf16_kernel(const float* __restrict__ in, __bf16* __restrict__ out, int n) {
  int i = blockIdx.x * blockDim.x + threadIdx.x;
  if (i < n) out[i] = (__bf16)in[i];
}

// Kernel 0b: zero the global sum/sumsq accumulators
__global__ void init_stats(float* stats) {
  if (threadIdx.x < 2) stats[threadIdx.x] = 0.0f;
}

// ---------------------------------------------------------------------------
// Kernel 1: qkv projection. z=0: x1 -> (q*norm, v1); z=1: x2 -> (k, v2).
// [2048 x 256] @ Wqv1^T [256 x 512], one 16x16 tile per wave, K=256 (8 WMMAs).
__global__ __launch_bounds__(32)
void proj_qkv(const float* __restrict__ x1, const float* __restrict__ x2,
              const __bf16* __restrict__ Wqv,
              __bf16* __restrict__ qB, __bf16* __restrict__ kB,
              __bf16* __restrict__ v1B, __bf16* __restrict__ v2B) {
  const int lane = threadIdx.x;
  const int mt = blockIdx.x;            // token tile (B*R / 16)
  const int ft = blockIdx.y;            // output-feature tile (512 / 16)
  const int zz = blockIdx.z;
  const float* x = zz ? x2 : x1;
  __bf16* out0 = zz ? kB : qB;          // s==0 half of qv
  __bf16* out1 = zz ? v2B : v1B;        // s==1 half of qv
  const float scale0 = zz ? 1.0f : 0.25f; // fold 1/sqrt(D) into q

  __builtin_prefetch((const char*)Wqv + (size_t)(ft * 16) * EE * 2 + lane * 256, 0, 0);

  v8f acc = {};
  #pragma unroll
  for (int kk = 0; kk < EE / 32; ++kk) {
    v16bf a, bfr;
    #pragma unroll
    for (int e = 0; e < 16; ++e) {
      int t = mt * 16 + (lane & 15);
      a[e] = (__bf16)x[t * EE + kk * 32 + a_k(lane, e)];
      int f = ft * 16 + (lane & 15);
      bfr[e] = Wqv[f * EE + kk * 32 + b_k(lane, e)];
    }
    acc = wmma_bf16(a, bfr, acc);
  }
  #pragma unroll
  for (int i = 0; i < 8; ++i) {
    int t = mt * 16 + c_row(lane, i);
    int f = ft * 16 + (lane & 15);
    int s = f >> 8, h = (f >> 4) & 15, d = f & 15;
    int bb = t / RR, r = t % RR;
    float v = acc[i];
    if (s == 0) v *= scale0;
    __bf16* dst = (s == 0) ? out0 : out1;
    dst[((bb * HH + h) * RR + r) * DD + d] = (__bf16)v;
  }
}

// ---------------------------------------------------------------------------
// Kernel 2: dot[b,h,r,c] = (q*norm) . k  — one WMMA per 16x16 tile, K=16 padded to 32.
__global__ __launch_bounds__(32)
void qk_dot(const __bf16* __restrict__ qB, const __bf16* __restrict__ kB,
            float* __restrict__ scores) {
  const int lane = threadIdx.x;
  const int bh = blockIdx.x;
  const int rt = blockIdx.y, ct = blockIdx.z;
  v16bf a, bfr;
  #pragma unroll
  for (int e = 0; e < 16; ++e) {
    int ka = a_k(lane, e);
    a[e] = (ka < DD) ? qB[(bh * RR + rt * 16 + (lane & 15)) * DD + ka] : (__bf16)0.0f;
    int kb = b_k(lane, e);
    bfr[e] = (kb < DD) ? kB[(bh * CC + ct * 16 + (lane & 15)) * DD + kb] : (__bf16)0.0f;
  }
  v8f acc = {};
  acc = wmma_bf16(a, bfr, acc);
  #pragma unroll
  for (int i = 0; i < 8; ++i) {
    int r = rt * 16 + c_row(lane, i);
    int c = ct * 16 + (lane & 15);
    scores[(bh * RR + r) * CC + c] = acc[i];
  }
}

// ---------------------------------------------------------------------------
// Kernel 3: MixedScoreFF, fused. Block = (b, r, 16 c's). M-tile = 16 (r,c) pairs,
// input features = interleave(dot_h, cost) (2H=32 == one bf16 WMMA K-step).
// lin1: 16 WMMAs (N=256), relu staged bf16 in LDS; lin2: 8 WMMAs (K=256, N=16 heads).
// Writes ms2 in-place over `scores` (block reads exactly what it writes) and
// accumulates global sum / sumsq for the std(ddof=1) normalization.
__global__ __launch_bounds__(32)
void mixed_score_mlp(float* __restrict__ scores, const float* __restrict__ cost,
                     const __bf16* __restrict__ W1, const __bf16* __restrict__ W2,
                     float* __restrict__ stats) {
  __shared__ __bf16 lds[16 * 256];
  const int lane = threadIdx.x;
  const int b = blockIdx.x / RR, r = blockIdx.x % RR;
  const int cbase = blockIdx.y * 16;

  // Warm bf16 weights into cache (global_prefetch_b8, one cacheline stride/lane)
  __builtin_prefetch((const char*)W1 + lane * 512, 0, 0);   // 16 KB W1
  __builtin_prefetch((const char*)W2 + lane * 256, 0, 0);   // 8 KB  W2

  v16bf a;
  #pragma unroll
  for (int e = 0; e < 16; ++e) {
    int c = cbase + (lane & 15);
    int f = a_k(lane, e);               // feature = h*2 + s
    int h = f >> 1;
    float v = (f & 1) ? cost[(b * RR + r) * CC + c]
                      : scores[((b * HH + h) * RR + r) * CC + c];
    a[e] = (__bf16)v;
  }
  #pragma unroll
  for (int jt = 0; jt < 16; ++jt) {     // lin1 + relu
    v16bf bfr;
    #pragma unroll
    for (int e = 0; e < 16; ++e) {
      int j = jt * 16 + (lane & 15);
      bfr[e] = W1[j * 32 + b_k(lane, e)];
    }
    v8f c8 = {};
    c8 = wmma_bf16(a, bfr, c8);
    #pragma unroll
    for (int i = 0; i < 8; ++i)
      lds[c_row(lane, i) * 256 + jt * 16 + (lane & 15)] = (__bf16)fmaxf(c8[i], 0.0f);
  }
  __syncthreads();
  v8f acc = {};                         // lin2
  #pragma unroll
  for (int kk = 0; kk < 8; ++kk) {
    v16bf a2, b2;
    #pragma unroll
    for (int e = 0; e < 16; ++e) {
      a2[e] = lds[(lane & 15) * 256 + kk * 32 + a_k(lane, e)];
      b2[e] = W2[(lane & 15) * 256 + kk * 32 + b_k(lane, e)];
    }
    acc = wmma_bf16(a2, b2, acc);
  }
  float s = 0.f, ss = 0.f;
  #pragma unroll
  for (int i = 0; i < 8; ++i) {
    int c = cbase + c_row(lane, i);
    int h = lane & 15;
    float v = acc[i];
    scores[((b * HH + h) * RR + r) * CC + c] = v;
    s += v; ss += v * v;
  }
  #pragma unroll
  for (int o = 16; o > 0; o >>= 1) { s += __shfl_xor(s, o, 32); ss += __shfl_xor(ss, o, 32); }
  if (lane == 0) { atomicAdd(&stats[0], s); atomicAdd(&stats[1], ss); }
}

// ---------------------------------------------------------------------------
// Kernel 4: global-std softmax (over c, or over r if transposed) + weights @ v.
// Block = (b,h, tile of 16 rows). exp() staged f32 in LDS; 1/rowsum folded into
// bf16 A conversion. AV: K=512 -> 16 WMMAs against v[b][h][k][d]. Heads stored bf16.
__global__ __launch_bounds__(32)
void softmax_av(const float* __restrict__ scores, const __bf16* __restrict__ vB,
                const float* __restrict__ stats, __bf16* __restrict__ outHeads,
                int transposed) {
  __shared__ float lde[16 * 512];
  const int lane = threadIdx.x;
  const int bh = blockIdx.x;
  const int b = bh / HH, h = bh % HH;
  const int tile = blockIdx.y;
  const int row = lane & 15;
  const int half = lane >> 4;

  const float n = (float)BB * HH * RR * CC;
  const float sm = stats[0], sq = stats[1];
  const float inv = rsqrtf((sq - sm * sm / n) / (n - 1.0f));

  const int gr = tile * 16 + row;
  float mx = -3.4e38f;
  for (int i = 0; i < 256; ++i) {       // load logits, row max
    int k = half * 256 + i;
    float v = transposed ? scores[(bh * RR + k) * CC + gr]
                         : scores[(bh * RR + gr) * CC + k];
    v *= inv;
    lde[row * 512 + k] = v;
    mx = fmaxf(mx, v);
  }
  mx = fmaxf(mx, __shfl_xor(mx, 16, 32));
  float sum = 0.f;
  for (int i = 0; i < 256; ++i) {       // exp + row sum
    int k = half * 256 + i;
    float e = __expf(lde[row * 512 + k] - mx);
    lde[row * 512 + k] = e;
    sum += e;
  }
  sum += __shfl_xor(sum, 16, 32);
  const float rs = 1.0f / sum;          // mask is all-true -> sum > 0
  __syncthreads();

  v8f acc = {};
  const int m = lane & 15;              // A row per lane == softmax row per lane
  #pragma unroll
  for (int kk = 0; kk < 16; ++kk) {
    v16bf a, bfr;
    #pragma unroll
    for (int e = 0; e < 16; ++e) {
      a[e] = (__bf16)(lde[m * 512 + kk * 32 + a_k(lane, e)] * rs);
      int kb = kk * 32 + b_k(lane, e);  // RR == CC: same index either way
      bfr[e] = vB[(bh * CC + kb) * DD + m];
    }
    acc = wmma_bf16(a, bfr, acc);
  }
  #pragma unroll
  for (int i = 0; i < 8; ++i) {
    int t = tile * 16 + c_row(lane, i);
    outHeads[(b * RR + t) * EE + h * DD + (lane & 15)] = (__bf16)acc[i];
  }
}

// ---------------------------------------------------------------------------
// Kernel 5: output projections. z=0: h1heads @ Wo1^T; z=1: h2heads @ Wo2^T.
// Both operands already bf16 -> pure load + WMMA loop, f32 result to d_out.
__global__ __launch_bounds__(32)
void out_proj(const __bf16* __restrict__ h1, const __bf16* __restrict__ h2,
              const __bf16* __restrict__ Wo1, const __bf16* __restrict__ Wo2,
              float* __restrict__ out) {
  const int lane = threadIdx.x;
  const int mt = blockIdx.x, ft = blockIdx.y, zz = blockIdx.z;
  const __bf16* in = zz ? h2 : h1;
  const __bf16* W = zz ? Wo2 : Wo1;
  float* o = out + (size_t)zz * (BB * RR * EE);

  __builtin_prefetch((const char*)W + (size_t)(ft * 16) * EE * 2 + lane * 256, 0, 0);

  v8f acc = {};
  #pragma unroll
  for (int kk = 0; kk < EE / 32; ++kk) {
    v16bf a, bfr;
    #pragma unroll
    for (int e = 0; e < 16; ++e) {
      int t = mt * 16 + (lane & 15);
      a[e] = in[t * EE + kk * 32 + a_k(lane, e)];
      int f = ft * 16 + (lane & 15);
      bfr[e] = W[f * EE + kk * 32 + b_k(lane, e)];
    }
    acc = wmma_bf16(a, bfr, acc);
  }
  #pragma unroll
  for (int i = 0; i < 8; ++i) {
    int t = mt * 16 + c_row(lane, i);
    int f = ft * 16 + (lane & 15);
    o[t * EE + f] = acc[i];
  }
}

// ---------------------------------------------------------------------------
extern "C" void kernel_launch(void* const* d_in, const int* in_sizes, int n_in,
                              void* d_out, int out_size, void* d_ws, size_t ws_size,
                              hipStream_t stream) {
  const float* x1   = (const float*)d_in[0];
  const float* x2   = (const float*)d_in[1];
  const float* cost = (const float*)d_in[2];
  // d_in[3] = attn_mask (all true) -> unused
  const float* Wqv1 = (const float*)d_in[4];
  const float* W1ms = (const float*)d_in[5];
  const float* W2ms = (const float*)d_in[6];
  const float* Wo1  = (const float*)d_in[7];
  const float* Wo2  = (const float*)d_in[8];

  // Workspace layout (~71 MB):
  // stats | qB kB v1B v2B (bf16) | scores f32 | h1h h2h (bf16) | bf16 weights
  char* ws = (char*)d_ws;
  size_t off = 0;
  float*  stats = (float*)(ws + off);  off += 256;
  __bf16* qB    = (__bf16*)(ws + off); off += (size_t)BB * HH * RR * DD * sizeof(__bf16);
  __bf16* kB    = (__bf16*)(ws + off); off += (size_t)BB * HH * CC * DD * sizeof(__bf16);
  __bf16* v1B   = (__bf16*)(ws + off); off += (size_t)BB * HH * RR * DD * sizeof(__bf16);
  __bf16* v2B   = (__bf16*)(ws + off); off += (size_t)BB * HH * CC * DD * sizeof(__bf16);
  float*  scr   = (float*)(ws + off);  off += (size_t)BB * HH * RR * CC * sizeof(float);
  __bf16* h1h   = (__bf16*)(ws + off); off += (size_t)BB * RR * EE * sizeof(__bf16);
  __bf16* h2h   = (__bf16*)(ws + off); off += (size_t)BB * CC * EE * sizeof(__bf16);
  __bf16* Wqvb  = (__bf16*)(ws + off); off += (size_t)2 * EE * EE * sizeof(__bf16);
  __bf16* W1b   = (__bf16*)(ws + off); off += (size_t)(HH * DD) * (2 * HH) * sizeof(__bf16);
  __bf16* W2b   = (__bf16*)(ws + off); off += (size_t)HH * (HH * DD) * sizeof(__bf16);
  __bf16* Wo1b  = (__bf16*)(ws + off); off += (size_t)EE * EE * sizeof(__bf16);
  __bf16* Wo2b  = (__bf16*)(ws + off); off += (size_t)EE * EE * sizeof(__bf16);
  float*  out   = (float*)d_out;

  // Weight pre-conversion (tiny; runs once per launch)
  f32_to_bf16_kernel<<<dim3((2 * EE * EE + 255) / 256), dim3(256), 0, stream>>>(Wqv1, Wqvb, 2 * EE * EE);
  f32_to_bf16_kernel<<<dim3((HH * DD * 2 * HH + 255) / 256), dim3(256), 0, stream>>>(W1ms, W1b, HH * DD * 2 * HH);
  f32_to_bf16_kernel<<<dim3((HH * HH * DD + 255) / 256), dim3(256), 0, stream>>>(W2ms, W2b, HH * HH * DD);
  f32_to_bf16_kernel<<<dim3((EE * EE + 255) / 256), dim3(256), 0, stream>>>(Wo1, Wo1b, EE * EE);
  f32_to_bf16_kernel<<<dim3((EE * EE + 255) / 256), dim3(256), 0, stream>>>(Wo2, Wo2b, EE * EE);
  init_stats<<<dim3(1), dim3(32), 0, stream>>>(stats);

  proj_qkv<<<dim3((BB * RR) / 16, (2 * EE) / 16, 2), dim3(32), 0, stream>>>(
      x1, x2, Wqvb, qB, kB, v1B, v2B);
  qk_dot<<<dim3(BB * HH, RR / 16, CC / 16), dim3(32), 0, stream>>>(qB, kB, scr);
  mixed_score_mlp<<<dim3(BB * RR, CC / 16), dim3(32), 0, stream>>>(
      scr, cost, W1b, W2b, stats);
  softmax_av<<<dim3(BB * HH, RR / 16), dim3(32), 0, stream>>>(scr, v2B, stats, h1h, 0);
  softmax_av<<<dim3(BB * HH, CC / 16), dim3(32), 0, stream>>>(scr, v1B, stats, h2h, 1);
  out_proj<<<dim3((BB * RR) / 16, EE / 16, 2), dim3(32), 0, stream>>>(
      h1h, h2h, Wo1b, Wo2b, out);
}
